// MoE_47717086658662
// MI455X (gfx1250) — matrix operand order repaired and checked by reading, not compiled, not run on hardware
//
#include <hip/hip_runtime.h>
#include <hip/hip_bf16.h>

// ---------------- problem constants (match reference) ----------------
#define TOK   8192      // B*S tokens
#define DDIM  1024      // D
#define NEXP  8         // E
#define TOPK  2         // K
#define HDIM  4096      // 4*D

// ---------------- types ----------------
typedef __attribute__((ext_vector_type(16))) __bf16     v16bf;
typedef __attribute__((ext_vector_type(8)))  float      v8f;
typedef __attribute__((ext_vector_type(4)))  float      f32x4;
typedef __attribute__((ext_vector_type(4)))  unsigned   u32x4;
typedef __attribute__((ext_vector_type(4)))  int        i32x4;

#define AS1 __attribute__((address_space(1)))
#define AS3 __attribute__((address_space(3)))

// Async global->LDS (CDNA5, tracked with ASYNCcnt); sync fallback if the
// builtin is unavailable in this toolchain.
#if __has_builtin(__builtin_amdgcn_global_load_async_to_lds_b128)
#define ASYNC_COPY16(gsrc, ldst)                                              \
    __builtin_amdgcn_global_load_async_to_lds_b128(                           \
        (AS1 i32x4*)(gsrc), (AS3 i32x4*)(ldst), 0, 0)
#define ASYNC_WAIT() asm volatile("s_wait_asynccnt 0x0" ::: "memory")
#else
#define ASYNC_COPY16(gsrc, ldst) (*(u32x4*)(ldst) = *(const u32x4*)(gsrc))
#define ASYNC_WAIT() ((void)0)
#endif

union FragAB {           // one wave32 WMMA A/B operand: 16 bf16 per lane
    v16bf v;
    u32x4 q[2];
};

__device__ __forceinline__ unsigned short f2bf(float f) {
    unsigned u = __builtin_bit_cast(unsigned, f);
    u += 0x7FFFu + ((u >> 16) & 1u);           // round-to-nearest-even
    return (unsigned short)(u >> 16);
}
__device__ __forceinline__ unsigned pack2bf(float lo, float hi) {
    return (unsigned)f2bf(lo) | ((unsigned)f2bf(hi) << 16);
}

// ---------------- utility kernels ----------------
__global__ void zero_i32(int* __restrict__ p, int n) {
    int i = blockIdx.x * blockDim.x + threadIdx.x;
    if (i < n) p[i] = 0;
}

// fp32 x -> bf16 x, 8 elements per thread
__global__ __launch_bounds__(256)
void cvt_x(const float* __restrict__ x, unsigned short* __restrict__ xb, int n8) {
    int i = blockIdx.x * 256 + threadIdx.x;
    if (i >= n8) return;
    f32x4 a = ((const f32x4*)x)[i * 2 + 0];
    f32x4 b = ((const f32x4*)x)[i * 2 + 1];
    u32x4 o;
    o.x = pack2bf(a[0], a[1]);
    o.y = pack2bf(a[2], a[3]);
    o.z = pack2bf(b[0], b[1]);
    o.w = pack2bf(b[2], b[3]);
    ((u32x4*)xb)[i] = o;
}

// ---------------- router: one wave32 per token ----------------
// Records per-expert gather lists (tok) AND per-token inverse map:
// pos[t*2+k] = (expert<<16)|slot, gq[t*2+k] = gate. The inverse map makes the
// final combine deterministic regardless of atomic slot-assignment order.
__global__ __launch_bounds__(256)
void moe_router(const float* __restrict__ x, const float* __restrict__ Wr,
                const float* __restrict__ br,
                int* __restrict__ counts, int* __restrict__ tok,
                int* __restrict__ pos, float* __restrict__ gq) {
    const int wave = threadIdx.x >> 5;
    const int lane = threadIdx.x & 31;
    const int t = blockIdx.x * 8 + wave;

    float p[NEXP];
#pragma unroll
    for (int e = 0; e < NEXP; ++e) p[e] = 0.0f;

    for (int d = lane; d < DDIM; d += 32) {
        float xv = x[(size_t)t * DDIM + d];
#pragma unroll
        for (int e = 0; e < NEXP; ++e) p[e] += xv * Wr[d * NEXP + e];
    }
#pragma unroll
    for (int off = 16; off >= 1; off >>= 1) {
#pragma unroll
        for (int e = 0; e < NEXP; ++e) p[e] += __shfl_xor(p[e], off, 32);
    }
    if (lane == 0) {
#pragma unroll
        for (int e = 0; e < NEXP; ++e) p[e] += br[e];
        int i0 = 0; float v0 = p[0];
#pragma unroll
        for (int e = 1; e < NEXP; ++e) if (p[e] > v0) { v0 = p[e]; i0 = e; }
        int i1 = -1; float v1 = -3.0e38f;
#pragma unroll
        for (int e = 0; e < NEXP; ++e) if (e != i0 && p[e] > v1) { v1 = p[e]; i1 = e; }
        float e1 = __expf(v1 - v0);               // v0 is the max
        float inv = __builtin_amdgcn_rcpf(1.0f + e1);
        int s0 = atomicAdd(&counts[i0], 1);
        tok[i0 * TOK + s0] = t;
        int s1 = atomicAdd(&counts[i1], 1);
        tok[i1 * TOK + s1] = t;
        pos[t * 2 + 0] = (i0 << 16) | s0;  gq[t * 2 + 0] = inv;
        pos[t * 2 + 1] = (i1 << 16) | s1;  gq[t * 2 + 1] = e1 * inv;
    }
}

__global__ void moe_prefix(const int* __restrict__ counts, int* __restrict__ bases) {
    if (threadIdx.x == 0 && blockIdx.x == 0) {
        int acc = 0;
        for (int e = 0; e < NEXP; ++e) { bases[e] = acc; acc += counts[e]; }
    }
}

// ---------------- WMMA fragment helpers (ISA 7.12.2 bf16 layouts) ----------------
__device__ __forceinline__ void load_a_frag(FragAB& f, const unsigned short* lds,
                                            int row0, int l, int hi) {
    const unsigned short* p = lds + (row0 + l) * 40 + hi * 8;
    f.q[0] = *(const u32x4*)(p);
    f.q[1] = *(const u32x4*)(p + 16);
}
__device__ __forceinline__ void load_b_frag(FragAB& f, const unsigned short* lds,
                                            int col0, int l, int hi) {
    const unsigned short* p = lds + (col0 + l) * 40 + hi * 8;
    f.q[0] = *(const u32x4*)(p);
    f.q[1] = *(const u32x4*)(p + 16);
}

// ---------------- pass A: h = silu(x*Wg+bg) * (x*Wu+bu), 128x64 tile ----------
__global__ __launch_bounds__(256)
void moe_gateup(const unsigned short* __restrict__ xb,
                const float* __restrict__ Wg, const float* __restrict__ bg,
                const float* __restrict__ Wu, const float* __restrict__ bu,
                const int* __restrict__ tok, const int* __restrict__ counts,
                const int* __restrict__ bases,
                unsigned short* __restrict__ hbuf) {
    const int e  = blockIdx.z;
    const int mb = blockIdx.y;            // 128-token slot tile
    const int hh = blockIdx.x * 64;       // 64-wide H tile
    const int cnt = counts[e];
    if (mb * 128 >= cnt) return;          // uniform exit: EXEC all-1 at WMMA
    const int base = bases[e];

    __shared__ __align__(16) unsigned short Alds[128 * 40];
    __shared__ __align__(16) unsigned short Bg[64 * 40];
    __shared__ __align__(16) unsigned short Bu[64 * 40];

    const int tid  = threadIdx.x;
    const int lane = tid & 31, wave = tid >> 5;
    const int wm = (wave >> 1) * 32, wn = (wave & 1) * 32;
    const int l = lane & 15, hi = lane >> 4;

    // ---- hoisted gather pointers for A staging (2 rows per thread) ----
    const int ar = tid >> 2, ach = tid & 3;
    int sa0 = mb * 128 + ar;       sa0 = sa0 < cnt ? sa0 : cnt - 1;
    int sa1 = mb * 128 + ar + 64;  sa1 = sa1 < cnt ? sa1 : cnt - 1;
    const unsigned short* asrc0 = xb + (size_t)tok[e * TOK + sa0] * DDIM + ach * 8;
    const unsigned short* asrc1 = xb + (size_t)tok[e * TOK + sa1] * DDIM + ach * 8;
    unsigned short* adst0 = &Alds[ar * 40 + ach * 8];
    unsigned short* adst1 = &Alds[(ar + 64) * 40 + ach * 8];

    // ---- hoisted B staging task: pair of k rows (kp), 4 columns (c4) ----
    const int kp = tid >> 4, c4 = tid & 15;
    const float* wgp = Wg + (size_t)e * DDIM * HDIM + (size_t)(2 * kp) * HDIM + hh + c4 * 4;
    const float* wup = Wu + (size_t)e * DDIM * HDIM + (size_t)(2 * kp) * HDIM + hh + c4 * 4;

    v8f ag[2][2], au[2][2];
#pragma unroll
    for (int i = 0; i < 2; ++i)
#pragma unroll
        for (int j = 0; j < 2; ++j) { ag[i][j] = (v8f)0.0f; au[i][j] = (v8f)0.0f; }

    for (int kk = 0; kk < DDIM; kk += 32) {
        // A tile: async bf16 global -> LDS (overlaps with B conversion below)
        ASYNC_COPY16(asrc0 + kk, adst0);
        ASYNC_COPY16(asrc1 + kk, adst1);
        // B tiles: fp32 -> bf16, packed pair stores ([N][K] layout)
        {
            const float* g0 = wgp + (size_t)kk * HDIM;
            const float* u0 = wup + (size_t)kk * HDIM;
            f32x4 ga = *(const f32x4*)(g0);
            f32x4 gb = *(const f32x4*)(g0 + HDIM);
            f32x4 ua = *(const f32x4*)(u0);
            f32x4 ub = *(const f32x4*)(u0 + HDIM);
            __builtin_prefetch(g0 + 32 * HDIM, 0, 0);   // next k-step strip
            __builtin_prefetch(u0 + 32 * HDIM, 0, 0);
#pragma unroll
            for (int j = 0; j < 4; ++j) {
                *(unsigned*)&Bg[(c4 * 4 + j) * 40 + 2 * kp] = pack2bf(ga[j], gb[j]);
                *(unsigned*)&Bu[(c4 * 4 + j) * 40 + 2 * kp] = pack2bf(ua[j], ub[j]);
            }
        }
        ASYNC_WAIT();
        __syncthreads();

        FragAB a[2], fg[2], fu[2];
#pragma unroll
        for (int mt = 0; mt < 2; ++mt) load_a_frag(a[mt], Alds, wm + mt * 16, l, hi);
#pragma unroll
        for (int nt = 0; nt < 2; ++nt) {
            load_b_frag(fg[nt], Bg, wn + nt * 16, l, hi);
            load_b_frag(fu[nt], Bu, wn + nt * 16, l, hi);
        }
#pragma unroll
        for (int mt = 0; mt < 2; ++mt)
#pragma unroll
            for (int nt = 0; nt < 2; ++nt) {
                ag[mt][nt] = __builtin_amdgcn_wmma_f32_16x16x32_bf16(
                    false, a[mt].v, false, fg[nt].v, (short)0, ag[mt][nt], false, false);
                au[mt][nt] = __builtin_amdgcn_wmma_f32_16x16x32_bf16(
                    false, a[mt].v, false, fu[nt].v, (short)0, au[mt][nt], false, false);
            }
        __syncthreads();
    }

    // ---- epilogue: silu(g)*u -> bf16 h rows (fast sigmoid via v_rcp_f32) ----
#pragma unroll
    for (int mt = 0; mt < 2; ++mt)
#pragma unroll
        for (int nt = 0; nt < 2; ++nt)
#pragma unroll
            for (int r = 0; r < 8; ++r) {
                int s = mb * 128 + wm + mt * 16 + hi * 8 + r;
                int col = hh + wn + nt * 16 + l;
                if (s < cnt) {
                    float g = ag[mt][nt][r] + bg[e * HDIM + col];
                    float u = au[mt][nt][r] + bu[e * HDIM + col];
                    float sig = __builtin_amdgcn_rcpf(1.0f + __expf(-g));
                    hbuf[(size_t)(base + s) * HDIM + col] = f2bf(g * sig * u);
                }
            }
}

// ---------------- pass B: ybuf[row] = h*Wd + bd (unscaled), 128x64 tile -------
__global__ __launch_bounds__(256)
void moe_down(const unsigned short* __restrict__ hbuf,
              const float* __restrict__ Wd, const float* __restrict__ bd,
              const int* __restrict__ counts, const int* __restrict__ bases,
              float* __restrict__ ybuf) {
    const int e  = blockIdx.z;
    const int mb = blockIdx.y;
    const int dd = blockIdx.x * 64;
    const int cnt = counts[e];
    if (mb * 128 >= cnt) return;
    const int base = bases[e];

    __shared__ __align__(16) unsigned short Alds[128 * 40];
    __shared__ __align__(16) unsigned short Bd[64 * 40];

    const int tid  = threadIdx.x;
    const int lane = tid & 31, wave = tid >> 5;
    const int wm = (wave >> 1) * 32, wn = (wave & 1) * 32;
    const int l = lane & 15, hi = lane >> 4;

    // hoisted A staging pointers (h rows are slot-contiguous; clamp tail)
    const int ar = tid >> 2, ach = tid & 3;
    int sa0 = mb * 128 + ar;       sa0 = sa0 < cnt ? sa0 : cnt - 1;
    int sa1 = mb * 128 + ar + 64;  sa1 = sa1 < cnt ? sa1 : cnt - 1;
    const unsigned short* asrc0 = hbuf + (size_t)(base + sa0) * HDIM + ach * 8;
    const unsigned short* asrc1 = hbuf + (size_t)(base + sa1) * HDIM + ach * 8;
    unsigned short* adst0 = &Alds[ar * 40 + ach * 8];
    unsigned short* adst1 = &Alds[(ar + 64) * 40 + ach * 8];

    const int kp = tid >> 4, c4 = tid & 15;
    const float* wdp = Wd + (size_t)e * HDIM * DDIM + (size_t)(2 * kp) * DDIM + dd + c4 * 4;

    v8f acc[2][2];
#pragma unroll
    for (int i = 0; i < 2; ++i)
#pragma unroll
        for (int j = 0; j < 2; ++j) acc[i][j] = (v8f)0.0f;

    for (int kk = 0; kk < HDIM; kk += 32) {
        ASYNC_COPY16(asrc0 + kk, adst0);
        ASYNC_COPY16(asrc1 + kk, adst1);
        {
            const float* w0 = wdp + (size_t)kk * DDIM;
            f32x4 wa = *(const f32x4*)(w0);
            f32x4 wb = *(const f32x4*)(w0 + DDIM);
            __builtin_prefetch(w0 + 32 * DDIM, 0, 0);
#pragma unroll
            for (int j = 0; j < 4; ++j)
                *(unsigned*)&Bd[(c4 * 4 + j) * 40 + 2 * kp] = pack2bf(wa[j], wb[j]);
        }
        ASYNC_WAIT();
        __syncthreads();

        FragAB a[2], b[2];
#pragma unroll
        for (int mt = 0; mt < 2; ++mt) load_a_frag(a[mt], Alds, wm + mt * 16, l, hi);
#pragma unroll
        for (int nt = 0; nt < 2; ++nt) load_b_frag(b[nt], Bd, wn + nt * 16, l, hi);
#pragma unroll
        for (int mt = 0; mt < 2; ++mt)
#pragma unroll
            for (int nt = 0; nt < 2; ++nt)
                acc[mt][nt] = __builtin_amdgcn_wmma_f32_16x16x32_bf16(
                    false, a[mt].v, false, b[nt].v, (short)0, acc[mt][nt], false, false);
        __syncthreads();
    }

#pragma unroll
    for (int mt = 0; mt < 2; ++mt)
#pragma unroll
        for (int nt = 0; nt < 2; ++nt)
#pragma unroll
            for (int r = 0; r < 8; ++r) {
                int s = mb * 128 + wm + mt * 16 + hi * 8 + r;
                int col = dd + wn + nt * 16 + l;
                if (s < cnt)
                    ybuf[(size_t)(base + s) * DDIM + col] = acc[mt][nt][r] + bd[e * DDIM + col];
            }
}

// ---------------- combine: out[t] = g0*y[row0] + g1*y[row1] (deterministic) ---
__global__ __launch_bounds__(256)
void moe_combine(const float* __restrict__ ybuf, const int* __restrict__ pos,
                 const float* __restrict__ gq, const int* __restrict__ bases,
                 float* __restrict__ out, int n4) {
    int i = blockIdx.x * 256 + threadIdx.x;
    if (i >= n4) return;
    int t = i / (DDIM / 4);
    int c = i % (DDIM / 4);
    int p0 = pos[t * 2 + 0], p1 = pos[t * 2 + 1];
    float g0 = gq[t * 2 + 0], g1 = gq[t * 2 + 1];
    size_t r0 = (size_t)(bases[p0 >> 16] + (p0 & 0xFFFF));
    size_t r1 = (size_t)(bases[p1 >> 16] + (p1 & 0xFFFF));
    f32x4 a = ((const f32x4*)(ybuf + r0 * DDIM))[c];
    f32x4 b = ((const f32x4*)(ybuf + r1 * DDIM))[c];
    ((f32x4*)(out + (size_t)t * DDIM))[c] = g0 * a + g1 * b;
}

// ---------------- host launcher ----------------
extern "C" void kernel_launch(void* const* d_in, const int* in_sizes, int n_in,
                              void* d_out, int out_size, void* d_ws, size_t ws_size,
                              hipStream_t stream) {
    (void)in_sizes; (void)n_in; (void)ws_size; (void)out_size;
    const float* x  = (const float*)d_in[0];
    const float* Wr = (const float*)d_in[1];
    const float* br = (const float*)d_in[2];
    const float* Wg = (const float*)d_in[3];
    const float* bg = (const float*)d_in[4];
    const float* Wu = (const float*)d_in[5];
    const float* bu = (const float*)d_in[6];
    const float* Wd = (const float*)d_in[7];
    const float* bd = (const float*)d_in[8];
    float* out = (float*)d_out;

    char* ws = (char*)d_ws;
    size_t off = 0;
    int*   counts = (int*)(ws + off);            off += 256;
    int*   bases  = (int*)(ws + off);            off += 256;
    int*   tok    = (int*)(ws + off);            off += (size_t)NEXP * TOK * 4;
    int*   pos    = (int*)(ws + off);            off += (size_t)TOK * TOPK * 4;
    float* gq     = (float*)(ws + off);          off += (size_t)TOK * TOPK * 4;
    unsigned short* xb   = (unsigned short*)(ws + off); off += (size_t)TOK * DDIM * 2;
    unsigned short* hbuf = (unsigned short*)(ws + off); off += (size_t)TOK * TOPK * HDIM * 2;
    float* ybuf = (float*)(ws + off);            off += (size_t)TOK * TOPK * DDIM * 4;

    zero_i32<<<1, 32, 0, stream>>>(counts, NEXP);
    cvt_x<<<(TOK * DDIM / 8 + 255) / 256, 256, 0, stream>>>(x, xb, TOK * DDIM / 8);
    moe_router<<<TOK / 8, 256, 0, stream>>>(x, Wr, br, counts, tok, pos, gq);
    moe_prefix<<<1, 1, 0, stream>>>(counts, bases);
    moe_gateup<<<dim3(HDIM / 64, TOK / 128, NEXP), 256, 0, stream>>>(
        xb, Wg, bg, Wu, bu, tok, counts, bases, hbuf);
    moe_down<<<dim3(DDIM / 64, TOK / 128, NEXP), 256, 0, stream>>>(
        hbuf, Wd, bd, counts, bases, ybuf);
    moe_combine<<<(TOK * DDIM / 4 + 255) / 256, 256, 0, stream>>>(
        ybuf, pos, gq, bases, out, TOK * DDIM / 4);
}
